// LocalNonLocal2D_32719060861010
// MI455X (gfx1250) — matrix operand order repaired and checked by reading, not compiled, not run on hardware
//
#include <hip/hip_runtime.h>
#include <hip/hip_bf16.h>

typedef __attribute__((ext_vector_type(16))) _Float16 v16h;
typedef __attribute__((ext_vector_type(8)))  float    v8f;

#define KW   7      // kernel size
#define PAD  3      // KW/2
#define TS   4      // 4x4 spatial tile -> 16 queries (WMMA M)
#define RG   10     // region dim = TS + KW - 1
#define NQ   100    // RG*RG valid region positions
#define NQP  112    // region padded to 7 WMMA n-tiles
#define CCH  128
#define HH   80
#define WW   80
#define NBAT 4

__global__ __launch_bounds__(128)
void lnl2d_kernel(const float* __restrict__ x,
                  const float* __restrict__ feat,
                  float* __restrict__ out,
                  float* __restrict__ pairwise)
{
    // f16 staging of the 10x10 halo region (rows padded w/ zeros to 128)
    __shared__ _Float16 Xr[128][CCH];   // x region      [q][c]
    __shared__ _Float16 Fr[128][CCH];   // feat region   [q][c]
    __shared__ float    S[NQP][16];     // raw scores    [q][p]  (q-major -> b128 stores)
    __shared__ _Float16 Wn[16][CCH];    // masked, normalized weights, K-padded

    const int tid  = threadIdx.x;
    const int lane = tid & 31;
    // Force the wave id into an SGPR so all WMMA control flow is scalar
    // (s_cbranch), keeping EXEC all-ones around every v_wmma (ISA 7.12 req).
    const int wv   = __builtin_amdgcn_readfirstlane(tid >> 5);
    const int half = lane >> 4;         // K-half selector for A/B fragments
    const int mrow = lane & 15;         // A row (M) / B,D column (N)

    const int blk = blockIdx.x;
    const int nb  = blk / ((HH / TS) * (WW / TS));
    const int rem = blk % ((HH / TS) * (WW / TS));
    const int h0  = (rem / (WW / TS)) * TS;
    const int w0  = (rem % (WW / TS)) * TS;

    // ---- stage halo regions (zero padded) from HBM -> LDS (f32 -> f16) ----
    for (int idx = tid; idx < 128 * CCH; idx += 128) {
        const int q = idx & 127;        // region position (fast dim: contiguous w)
        const int c = idx >> 7;         // channel
        float xv = 0.f, fv = 0.f;
        if (q < NQ) {
            const int qy = q / RG, qx = q % RG;
            const int h = h0 - PAD + qy;
            const int w = w0 - PAD + qx;
            if (h >= 0 && h < HH && w >= 0 && w < WW) {
                const long off = (((long)nb * CCH + c) * HH + h) * WW + w;
                xv = x[off];
                fv = feat[off];
            }
        }
        Xr[q][c] = (_Float16)xv;
        Fr[q][c] = (_Float16)fv;
    }
    for (int idx = tid; idx < 16 * CCH; idx += 128)
        ((_Float16*)Wn)[idx] = (_Float16)0.f;
    __syncthreads();

    // A-fragment K index for packed element e (16x32 f16 A layout, ISA 7.12.2)
    auto kIdxA = [&](int e) {
        const int v = e >> 1, j = e & 1;
        return (v < 4) ? (half * 8 + 2 * v + j)
                       : (16 + half * 8 + 2 * (v - 4) + j);
    };

    // ---- GEMM1: S[q][p] = sum_c Xq[p,c] * Xr[q,c] (7 n-tiles over region) ----
    // A (query fragment) is jn-invariant: hoist all 4 K-chunk fragments.
    v16h aq[4];
#pragma unroll
    for (int kc = 0; kc < 4; ++kc) {
        const int qa = ((mrow >> 2) + PAD) * RG + (mrow & 3) + PAD;
#pragma unroll
        for (int e = 0; e < 16; ++e)
            aq[kc][e] = Xr[qa][kc * 32 + kIdxA(e)];
    }
    for (int jn = wv; jn < 7; jn += 4) {             // scalar (SGPR) loop
        v8f acc = {0.f, 0.f, 0.f, 0.f, 0.f, 0.f, 0.f, 0.f};
#pragma unroll
        for (int kc = 0; kc < 4; ++kc) {
            const int kofs = kc * 32;
            v16h b;
#pragma unroll
            for (int e = 0; e < 16; ++e) {           // B: (k=c, n=q) = Xr[q][c]
                const int v = e >> 1, j = e & 1;
                const int k = (v < 4) ? (half * 8 + 2 * v + j) : (16 + half * 8 + 2 * (v - 4) + j);
                b[e] = Xr[16 * jn + mrow][kofs + k];
            }
            acc = __builtin_amdgcn_wmma_f32_16x16x32_f16(false, aq[kc], false, b,
                                                         (short)0, acc, false, false);
        }
        // lane's 8 rows are contiguous at fixed column -> ds_store_b128 x2
        float* sdst = &S[16 * jn + mrow][half * 8];
#pragma unroll
        for (int rr = 0; rr < 8; ++rr)
            sdst[rr] = acc[rr];
    }
    __syncthreads();

    // ---- mask to 7x7 window, normalize by plain sum, emit pairwise output ----
    if (tid < 16) {
        const int p = tid, py = p >> 2, px = p & 3;
        float vals[KW * KW];
        float sum = 0.f;
#pragma unroll
        for (int i = 0; i < KW; ++i)
#pragma unroll
            for (int j = 0; j < KW; ++j) {
                const float v = S[(py + i) * RG + (px + j)][p];
                vals[i * KW + j] = v;
                sum += v;
            }
        const float inv = 1.0f / sum;                // reference: no epsilon
        float* pw = pairwise +
                    ((long)((nb * HH + (h0 + py)) * WW + (w0 + px))) * (KW * KW);
#pragma unroll
        for (int i = 0; i < KW; ++i)
#pragma unroll
            for (int j = 0; j < KW; ++j) {
                const float wn = vals[i * KW + j] * inv;
                Wn[p][(py + i) * RG + (px + j)] = (_Float16)wn;
                pw[i * KW + j] = wn;
            }
    }
    __syncthreads();

    // ---- GEMM2: out[p][c] = sum_q Wn[p][q] * Fr[q][c] (8 n-tiles over C) ----
    auto gemm2_tile = [&](int jn) {
        const int c0 = 16 * jn;
        v8f acc = {0.f, 0.f, 0.f, 0.f, 0.f, 0.f, 0.f, 0.f};
#pragma unroll
        for (int kc = 0; kc < 4; ++kc) {
            const int kofs = kc * 32;
            v16h a, b;
#pragma unroll
            for (int e = 0; e < 16; ++e)             // A: masked weights
                a[e] = Wn[mrow][kofs + kIdxA(e)];
#pragma unroll
            for (int e = 0; e < 16; ++e) {           // B: (k=q, n=c) = Fr[q][c]
                const int v = e >> 1, j = e & 1;
                const int k = (v < 4) ? (half * 8 + 2 * v + j) : (16 + half * 8 + 2 * (v - 4) + j);
                b[e] = Fr[kofs + k][c0 + mrow];
            }
            acc = __builtin_amdgcn_wmma_f32_16x16x32_f16(false, a, false, b,
                                                         (short)0, acc, false, false);
        }
#pragma unroll
        for (int rr = 0; rr < 8; ++rr) {
            const int p = half * 8 + rr;             // query position
            const int c = c0 + mrow;
            const int h = h0 + (p >> 2), w = w0 + (p & 3);
            out[(((long)nb * CCH + c) * HH + h) * WW + w] = acc[rr];
        }
    };
    gemm2_tile(2 * wv);          // wv is SGPR -> no exec masking
    gemm2_tile(2 * wv + 1);
}

extern "C" void kernel_launch(void* const* d_in, const int* in_sizes, int n_in,
                              void* d_out, int out_size, void* d_ws, size_t ws_size,
                              hipStream_t stream) {
    const float* x    = (const float*)d_in[0];
    const float* feat = (const float*)d_in[1];
    float* out      = (float*)d_out;
    float* pairwise = out + (long)NBAT * CCH * HH * WW;  // outputs concatenated

    const int nblocks = NBAT * (HH / TS) * (WW / TS);    // 1600 tiles
    lnl2d_kernel<<<nblocks, 128, 0, stream>>>(x, feat, out, pairwise);
}
